// InfoNCELoss_46514495815979
// MI455X (gfx1250) — matrix operand order repaired and checked by reading, not compiled, not run on hardware
//
#include <hip/hip_runtime.h>
#include <stdint.h>

typedef __attribute__((ext_vector_type(16))) __bf16 v16bf;
typedef __attribute__((ext_vector_type(8)))  float  v8f;

#define B_ROWS   8192
#define DIM      512
#define NSPLIT   8
#define TEMP_INV (1.0f / 0.07f)

// ---------- bf16 helpers (bit-level, round-to-nearest-even) ----------
__device__ __forceinline__ uint16_t f32_to_bf16_rne(float f) {
  unsigned int u = __float_as_uint(f);
  u += 0x7FFFu + ((u >> 16) & 1u);
  return (uint16_t)(u >> 16);
}
__device__ __forceinline__ float bf16_to_f32(uint16_t h) {
  return __uint_as_float(((unsigned int)h) << 16);
}

// ---------- Kernel A: row-normalize, split into bf16 hi/lo ----------
__global__ __launch_bounds__(256) void nrm_split_kernel(
    const float* __restrict__ x,
    uint16_t* __restrict__ ahi, uint16_t* __restrict__ alo) {
  __shared__ float red[256];
  const int row = blockIdx.x;
  const int tid = threadIdx.x;
  const float* xr = x + (size_t)row * DIM;
  float a = xr[tid];
  float b = xr[tid + 256];
  red[tid] = a * a + b * b;
  __syncthreads();
  for (int off = 128; off > 0; off >>= 1) {
    if (tid < off) red[tid] += red[tid + off];
    __syncthreads();
  }
  float nrm = fmaxf(sqrtf(red[0]), 1e-12f);
  float inv = 1.0f / nrm;
  float fa = a * inv, fb = b * inv;
  uint16_t ha = f32_to_bf16_rne(fa);
  uint16_t hb = f32_to_bf16_rne(fb);
  uint16_t la = f32_to_bf16_rne(fa - bf16_to_f32(ha));
  uint16_t lb = f32_to_bf16_rne(fb - bf16_to_f32(hb));
  const size_t base = (size_t)row * DIM;
  ahi[base + tid] = ha;  ahi[base + tid + 256] = hb;
  alo[base + tid] = la;  alo[base + tid + 256] = lb;
}

// ---------- fragment loaders (two 16B chunks, gap in elements) ----------
__device__ __forceinline__ v16bf load_frag(const uint16_t* __restrict__ p, int gapElems) {
  union { uint4 u[2]; v16bf v; } r;
  r.u[0] = *reinterpret_cast<const uint4*>(p);
  r.u[1] = *reinterpret_cast<const uint4*>(p + gapElems);
  return r.v;
}

#define WMMA_BF16(A, Bm, C) \
  __builtin_amdgcn_wmma_f32_16x16x32_bf16(false, (A), false, (Bm), (short)0, (C), false, false)

#define TILE_EPILOGUE(CV, TOFF)                                      \
  {                                                                  \
    const int jcol = jb + (TOFF) + n;                                \
    _Pragma("unroll")                                                \
    for (int v = 0; v < 8; ++v) {                                    \
      float val = (CV)[v] * TEMP_INV;                                \
      tp[v] += (jcol == cy[v]) ? val : 0.0f;                         \
      val = (jcol == irow[v]) ? -__builtin_inff() : val;             \
      float nm = fmaxf(m[v], val);                                   \
      s[v] = s[v] * __expf(m[v] - nm) + __expf(val - nm);            \
      m[v] = nm;                                                     \
    }                                                                \
  }

// ---------- Kernel B: fused F*F^T (split-bf16 WMMA) + online logsumexp ----------
__global__ __launch_bounds__(256) void infonce_main_kernel(
    const uint16_t* __restrict__ ahi, const uint16_t* __restrict__ alo,
    const int* __restrict__ y,
    float* __restrict__ m_part, float* __restrict__ s_part,
    float* __restrict__ t_part) {
  const int lane = threadIdx.x & 31;
  const int wave = threadIdx.x >> 5;     // split id: 0..NSPLIT-1
  const int half = lane >> 4;            // 0 or 1
  const int n    = lane & 15;
  const int ibase  = blockIdx.x * 16;
  const int jstart = wave * (B_ROWS / NSPLIT);

  // A fragment addressing: lane holds row = ibase+n, chunks at k0 and k0+8
  const uint16_t* arow_hi = ahi + (size_t)(ibase + n) * DIM;
  const uint16_t* arow_lo = alo + (size_t)(ibase + n) * DIM;
  const int ak0 = half * 16;   // A chunk base (contiguous 16 elems)
  const int bk0 = half * 16;   // B chunk base (contiguous 16 elems)

  float m[8], s[8], tp[8];
  int   cy[8], irow[8];
#pragma unroll
  for (int v = 0; v < 8; ++v) {
    const int i = ibase + v + 8 * half;   // C row held by this lane in VGPR v
    irow[v] = i;
    const int yv = y[i];
    cy[v] = yv + ((yv >= i) ? 1 : 0);
    m[v] = -1e30f; s[v] = 0.0f; tp[v] = 0.0f;
  }

  for (int jb = jstart; jb < jstart + (B_ROWS / NSPLIT); jb += 64) {
    v8f c0 = {}, c1 = {}, c2 = {}, c3 = {};
    // B = F^T: column (jb + t*16 + n) of B is row (jb + t*16 + n) of F
    const uint16_t* b0h = ahi + (size_t)(jb +  0 + n) * DIM;
    const uint16_t* b1h = ahi + (size_t)(jb + 16 + n) * DIM;
    const uint16_t* b2h = ahi + (size_t)(jb + 32 + n) * DIM;
    const uint16_t* b3h = ahi + (size_t)(jb + 48 + n) * DIM;
    const uint16_t* b0l = alo + (size_t)(jb +  0 + n) * DIM;
    const uint16_t* b1l = alo + (size_t)(jb + 16 + n) * DIM;
    const uint16_t* b2l = alo + (size_t)(jb + 32 + n) * DIM;
    const uint16_t* b3l = alo + (size_t)(jb + 48 + n) * DIM;

    for (int kc = 0; kc < DIM; kc += 32) {
      const v16bf a_h = load_frag(arow_hi + kc + ak0, 8);
      const v16bf a_l = load_frag(arow_lo + kc + ak0, 8);

      v16bf bh, bl;
      bh = load_frag(b0h + kc + bk0, 8);  bl = load_frag(b0l + kc + bk0, 8);
      c0 = WMMA_BF16(a_h, bh, c0);  c0 = WMMA_BF16(a_h, bl, c0);  c0 = WMMA_BF16(a_l, bh, c0);
      bh = load_frag(b1h + kc + bk0, 8);  bl = load_frag(b1l + kc + bk0, 8);
      c1 = WMMA_BF16(a_h, bh, c1);  c1 = WMMA_BF16(a_h, bl, c1);  c1 = WMMA_BF16(a_l, bh, c1);
      bh = load_frag(b2h + kc + bk0, 8);  bl = load_frag(b2l + kc + bk0, 8);
      c2 = WMMA_BF16(a_h, bh, c2);  c2 = WMMA_BF16(a_h, bl, c2);  c2 = WMMA_BF16(a_l, bh, c2);
      bh = load_frag(b3h + kc + bk0, 8);  bl = load_frag(b3l + kc + bk0, 8);
      c3 = WMMA_BF16(a_h, bh, c3);  c3 = WMMA_BF16(a_h, bl, c3);  c3 = WMMA_BF16(a_l, bh, c3);
    }

    TILE_EPILOGUE(c0,  0)
    TILE_EPILOGUE(c1, 16)
    TILE_EPILOGUE(c2, 32)
    TILE_EPILOGUE(c3, 48)
  }

  // Butterfly-merge the 16 column-classes (stays within each 16-lane half).
#pragma unroll
  for (int mask = 1; mask <= 8; mask <<= 1) {
#pragma unroll
    for (int v = 0; v < 8; ++v) {
      const float mo = __shfl_xor(m[v],  mask, 32);
      const float so = __shfl_xor(s[v],  mask, 32);
      const float to = __shfl_xor(tp[v], mask, 32);
      const float nm = fmaxf(m[v], mo);
      s[v] = s[v] * __expf(m[v] - nm) + so * __expf(mo - nm);
      m[v] = nm;
      tp[v] += to;
    }
  }

  if (n == 0) {   // lane 0 -> rows 0..7, lane 16 -> rows 8..15
#pragma unroll
    for (int v = 0; v < 8; ++v) {
      const size_t idx = (size_t)irow[v] * NSPLIT + wave;
      m_part[idx] = m[v];
      s_part[idx] = s[v];
      t_part[idx] = tp[v];
    }
  }
}

// ---------- Kernel C: merge splits, mean NLL ----------
__global__ __launch_bounds__(256) void finalize_kernel(
    const float* __restrict__ m_part, const float* __restrict__ s_part,
    const float* __restrict__ t_part, float* __restrict__ out) {
  __shared__ float red[256];
  const int tid = threadIdx.x;
  float acc = 0.0f;
  for (int row = tid; row < B_ROWS; row += 256) {
    float M = -1e30f;
#pragma unroll
    for (int p = 0; p < NSPLIT; ++p) M = fmaxf(M, m_part[(size_t)row * NSPLIT + p]);
    float S = 0.0f, T = 0.0f;
#pragma unroll
    for (int p = 0; p < NSPLIT; ++p) {
      S += s_part[(size_t)row * NSPLIT + p] * __expf(m_part[(size_t)row * NSPLIT + p] - M);
      T += t_part[(size_t)row * NSPLIT + p];
    }
    acc += (M + __logf(S)) - T;
  }
  red[tid] = acc;
  __syncthreads();
  for (int off = 128; off > 0; off >>= 1) {
    if (tid < off) red[tid] += red[tid + off];
    __syncthreads();
  }
  if (tid == 0) out[0] = red[0] / (float)B_ROWS;
}

extern "C" void kernel_launch(void* const* d_in, const int* in_sizes, int n_in,
                              void* d_out, int out_size, void* d_ws, size_t ws_size,
                              hipStream_t stream) {
  const float* x = (const float*)d_in[0];
  const int*   y = (const int*)d_in[1];

  char* ws = (char*)d_ws;
  size_t off = 0;
  uint16_t* ahi = (uint16_t*)(ws + off); off += (size_t)B_ROWS * DIM * sizeof(uint16_t);
  uint16_t* alo = (uint16_t*)(ws + off); off += (size_t)B_ROWS * DIM * sizeof(uint16_t);
  float* m_part = (float*)(ws + off);    off += (size_t)B_ROWS * NSPLIT * sizeof(float);
  float* s_part = (float*)(ws + off);    off += (size_t)B_ROWS * NSPLIT * sizeof(float);
  float* t_part = (float*)(ws + off);    off += (size_t)B_ROWS * NSPLIT * sizeof(float);
  (void)ws_size; (void)in_sizes; (void)n_in; (void)out_size;

  nrm_split_kernel<<<B_ROWS, 256, 0, stream>>>(x, ahi, alo);
  infonce_main_kernel<<<B_ROWS / 16, 256, 0, stream>>>(ahi, alo, y,
                                                       m_part, s_part, t_part);
  finalize_kernel<<<1, 256, 0, stream>>>(m_part, s_part, t_part, (float*)d_out);
}
